// MultiHeadAttention_51049981280848
// MI455X (gfx1250) — matrix operand order repaired and checked by reading, not compiled, not run on hardware
//
#include <hip/hip_runtime.h>

typedef __attribute__((ext_vector_type(16))) __bf16 v16bf;
typedef __attribute__((ext_vector_type(8)))  __bf16 v8bf;
typedef __attribute__((ext_vector_type(8)))  float  v8f;
typedef __attribute__((ext_vector_type(4)))  unsigned int uint32x4;
typedef __attribute__((ext_vector_type(4)))  int int32x4;
typedef __attribute__((ext_vector_type(8)))  int int32x8;

#define HIDDEN 1024
#define NHEADS 16
#define HDIM   64
#define SEQ    2048
#define BATCH  2
#define MROWS  (BATCH*SEQ)   /* 4096 */

#if __has_builtin(__builtin_amdgcn_tensor_load_to_lds) && __has_builtin(__builtin_amdgcn_s_wait_tensorcnt)
#define USE_TDM 1
#else
#define USE_TDM 0
#endif

static __device__ __forceinline__ v8f wmma_bf16(v16bf a, v16bf b, v8f c) {
  return __builtin_amdgcn_wmma_f32_16x16x32_bf16(false, a, false, b, (short)0, c, false, false);
}

// A-fragment (16x32 bf16, row-major source). Lane: row m = lane%16,
// 16 contiguous K values at K = h*16 (matches B-fragment order)
static __device__ __forceinline__ v16bf load_a_frag(const __bf16* __restrict__ A, int lda, int kk) {
  const int lane = threadIdx.x & 31;
  const int m = lane & 15, hl = lane >> 4;
  const __bf16* p = A + m * lda + kk + hl * 16;
  v8bf r0 = *(const v8bf*)(p);
  v8bf r1 = *(const v8bf*)(p + 8);
  v16bf r;
#pragma unroll
  for (int i = 0; i < 8; ++i) { r[i] = r0[i]; r[8 + i] = r1[i]; }
  return r;
}

// B-fragment (32x16 bf16) from N-major source (Bt[n][k]): lane = column
// n = lane%16, 16 contiguous K values at K = h*16.
static __device__ __forceinline__ v16bf load_b_frag(const __bf16* __restrict__ Bt, int ldb, int kk) {
  const int lane = threadIdx.x & 31;
  const int n = lane & 15, hl = lane >> 4;
  return *(const v16bf*)(Bt + n * ldb + kk + hl * 16);
}

__global__ __launch_bounds__(256)
void cast_bf16_kernel(const float* __restrict__ in, __bf16* __restrict__ out, int n) {
  int i = blockIdx.x * 256 + threadIdx.x;
  if (i < n) out[i] = (__bf16)in[i];
}

// W[k][n] fp32 (1024x1024) -> Wt[n][k] bf16
__global__ __launch_bounds__(256)
void transpose_cast_kernel(const float* __restrict__ W, __bf16* __restrict__ Wt) {
  int i = blockIdx.x * 256 + threadIdx.x;
  int k = i >> 10, n = i & 1023;
  Wt[n * 1024 + k] = (__bf16)W[i];
}

// C = A(bf16,[M x 1024]) @ Bt(bf16, N-major [1024 x 1024]) + bias.
// Block = 4 waves = 64(M) x 64(N) tile. B tile (64n x 64k, 8KB) staged in LDS
// by the Tensor Data Mover (double buffered, TENSORcnt + barrier sync). The
// TDM descriptor requests 16B of LDS padding per 128B row -> 144B pitch:
// lane n's row starts at n*36 dwords, distinct mod 64 banks -> conflict-free
// ds_load_b128 B-fragments. 8 WMMAs per sync point.
// store_mode: 0 = [B][H][S][D] bf16 (Q/K), 1 = [B][H][D][S] bf16 (V^T),
//             2 = fp32 row-major (final output)
__global__ __launch_bounds__(128)
void gemm_bf16_kernel(const __bf16* __restrict__ A, const __bf16* __restrict__ Bt,
                      const float* __restrict__ bias, void* __restrict__ Cout,
                      int mblocks, int store_mode)
{
  __shared__ __align__(16) __bf16 bbuf[2][64 * 72];   // 64 rows, pitch 72 elems (144B)
  const int w    = threadIdx.x >> 5;
  const int lane = threadIdx.x & 31;
  const int n = lane & 15, hl = lane >> 4;
  const int mb = blockIdx.x % mblocks;       // 64-row block
  const int nt = blockIdx.x / mblocks;
  const int n0 = nt * 64;
  const int mt = mb * 4 + w;                 // this wave's 16-row tile
  const __bf16* Arow = A + (long)mt * 16 * HIDDEN;
  const int NSTAGE = HIDDEN / 64;            // 16

#if USE_TDM
  auto issue_tdm = [&](int buf, int k0) {
    unsigned long long ga = (unsigned long long)(Bt + (long)n0 * HIDDEN + k0);
    // flat->LDS aperture truncation: low 32 bits of a __shared__ flat pointer
    // are the LDS byte offset (ISA 10.2 aperture mapping)
    unsigned lds = (unsigned)(unsigned long long)(void*)&bbuf[buf][0];
    uint32x4 g0;
    g0.x = 1u;                                   // count=1, user descriptor
    g0.y = lds;                                  // lds_addr
    g0.z = (unsigned)(ga & 0xffffffffu);         // global_addr[31:0]
    g0.w = (unsigned)(ga >> 32) | (2u << 30);    // global_addr[56:32] | type=2
    int32x8 g1;
    g1[0] = 0x07110000;   // data_size=1(2B) | pad_en | pad_interval=4(128B) | pad_amount=3(16B)
    g1[1] = 0x04000000;   // tensor_dim0 = 1024 (low16 in [31:16])
    g1[2] = 0;            // tensor_dim0 hi / tensor_dim1 lo
    g1[3] = 0x00400010;   // tensor_dim1 = 1<<20 (hi16=16) | tile_dim0 = 64
    g1[4] = 64;           // tile_dim1 = 64 rows
    g1[5] = 1024;         // tensor_dim0_stride = 1024 elements
    g1[6] = 0;
    g1[7] = 0;
    int32x4 z4 = {0, 0, 0, 0};
#if __clang_major__ >= 23
    int32x8 z8 = {0, 0, 0, 0, 0, 0, 0, 0};
    __builtin_amdgcn_tensor_load_to_lds(g0, g1, z4, z4, z8, 0);
#else
    __builtin_amdgcn_tensor_load_to_lds(g0, g1, z4, z4, 0);
#endif
  };
  if (w == 0) issue_tdm(0, 0);
#endif

  v8f acc[4] = {};
  v16bf a0 = load_a_frag(Arow, HIDDEN, 0);
  v16bf a1 = load_a_frag(Arow, HIDDEN, 32);

  for (int s = 0; s < NSTAGE; ++s) {
    const int k0 = s * 64;
#if USE_TDM
    if (w == 0) __builtin_amdgcn_s_wait_tensorcnt(0);   // buf[s&1] landed
    __syncthreads();                                     // publish to all waves
    if (w == 0 && s + 1 < NSTAGE) issue_tdm((s + 1) & 1, k0 + 64);
    const __bf16* bb = &bbuf[s & 1][0];
#endif
    // software pipeline A fragments: next stage's global loads overlap WMMAs
    v16bf a0n = a0, a1n = a1;
    if (s + 1 < NSTAGE) {
      a0n = load_a_frag(Arow, HIDDEN, k0 + 64);
      a1n = load_a_frag(Arow, HIDDEN, k0 + 96);
    }
#pragma unroll
    for (int kk = 0; kk < 2; ++kk) {
      v16bf a = (kk == 0) ? a0 : a1;
#pragma unroll
      for (int t = 0; t < 4; ++t) {
#if USE_TDM
        v16bf b = *(const v16bf*)(bb + (t * 16 + n) * 72 + kk * 32 + hl * 16);
#else
        v16bf b = load_b_frag(Bt + (long)(n0 + t * 16) * HIDDEN, HIDDEN, k0 + kk * 32);
#endif
        acc[t] = wmma_bf16(a, b, acc[t]);
      }
    }
#if USE_TDM
    __syncthreads();                 // all readers done before buffer reuse
#endif
    a0 = a0n; a1 = a1n;
  }

  if (store_mode == 2) {
    float* out = (float*)Cout;
#pragma unroll
    for (int t = 0; t < 4; ++t) {
      int ng = n0 + t * 16 + n;
      float bv = bias[ng];
#pragma unroll
      for (int vi = 0; vi < 8; ++vi) {
        int row = mt * 16 + hl * 8 + vi;
        out[(long)row * HIDDEN + ng] = acc[t][vi] + bv;
      }
    }
  } else {
    __bf16* out = (__bf16*)Cout;
#pragma unroll
    for (int t = 0; t < 4; ++t) {
      int ng = n0 + t * 16 + n;
      int head = ng >> 6, d = ng & 63;
      float bv = bias[ng];
#pragma unroll
      for (int vi = 0; vi < 8; ++vi) {
        int row = mt * 16 + hl * 8 + vi;
        int bb2 = row >> 11, sr = row & (SEQ - 1);
        float v = acc[t][vi] + bv;
        long idx;
        if (store_mode == 0) idx = (((long)(bb2 * NHEADS + head) * SEQ)  + sr) * HDIM + d;
        else                 idx = (((long)(bb2 * NHEADS + head) * HDIM) + d ) * SEQ  + sr;
        out[idx] = (__bf16)v;
      }
    }
  }
}

// Flash attention: 1 block per (b, head, 64-query tile); 4 waves, each owns
// 16 query rows. Keys streamed in chunks of 32 with online softmax.
// Register-pipelined: K-fragments/mask for chunk i+1 prefetch under chunk i's
// softmax; V fragments prefetch before the softmax VALU section.
__global__ __launch_bounds__(128)
void attn_kernel(const __bf16* __restrict__ Q, const __bf16* __restrict__ K,
                 const __bf16* __restrict__ Vt, const float* __restrict__ mask,
                 __bf16* __restrict__ Ctx)
{
  __shared__ __align__(16) __bf16 lds_p[4][16 * 40];  // 16x32 P tile/wave, pad 8
  const int w    = threadIdx.x >> 5;
  const int lane = threadIdx.x & 31;
  const int n = lane & 15, hl = lane >> 4;
  const int b = blockIdx.z, head = blockIdx.y;
  const int q0 = blockIdx.x * 64 + w * 16;
  const __bf16* Qh = Q  + (long)(b * NHEADS + head) * SEQ * HDIM;
  const __bf16* Kh = K  + (long)(b * NHEADS + head) * SEQ * HDIM;
  const __bf16* Vh = Vt + (long)(b * NHEADS + head) * HDIM * SEQ;

  v16bf qf0 = load_a_frag(Qh + (long)q0 * HDIM, HDIM, 0);
  v16bf qf1 = load_a_frag(Qh + (long)q0 * HDIM, HDIM, 32);

  v8f acc[4] = {};
  float rmax[8], rsum[8];
#pragma unroll
  for (int i = 0; i < 8; ++i) { rmax[i] = -3.0e38f; rsum[i] = 0.0f; }

  // prologue: fragments for chunk 0
  v16bf kb[2][2];
  float mv[2];
#pragma unroll
  for (int j = 0; j < 2; ++j) {
    const __bf16* Kp = Kh + (long)(j * 16) * HDIM;
    kb[j][0] = load_b_frag(Kp, HDIM, 0);
    kb[j][1] = load_b_frag(Kp, HDIM, 32);
    mv[j] = mask[b * SEQ + j * 16 + n];
  }

  for (int k0 = 0; k0 < SEQ; k0 += 32) {
    // scores for this chunk from prefetched K fragments
    v8f s[2];
    float addm[2];
#pragma unroll
    for (int j = 0; j < 2; ++j) {
      v8f z = {};
      z = wmma_bf16(qf0, kb[j][0], z);
      z = wmma_bf16(qf1, kb[j][1], z);
      s[j] = z;
      addm[j] = (1.0f - mv[j]) * -10000.0f;
    }
    // prefetch V fragments for this chunk (consumed after softmax + LDS trip)
    v16bf vf[4];
#pragma unroll
    for (int t = 0; t < 4; ++t)
      vf[t] = *(const v16bf*)(Vh + (long)(t * 16 + n) * SEQ + k0 + hl * 16);
    // prefetch K fragments + mask for the next chunk
    v16bf kbn[2][2];
    float mvn[2];
    if (k0 + 32 < SEQ) {
#pragma unroll
      for (int j = 0; j < 2; ++j) {
        const __bf16* Kp = Kh + (long)(k0 + 32 + j * 16) * HDIM;
        kbn[j][0] = load_b_frag(Kp, HDIM, 0);
        kbn[j][1] = load_b_frag(Kp, HDIM, 32);
        mvn[j] = mask[b * SEQ + k0 + 32 + j * 16 + n];
      }
    } else {
#pragma unroll
      for (int j = 0; j < 2; ++j) { kbn[j][0] = kb[j][0]; kbn[j][1] = kb[j][1]; mvn[j] = mv[j]; }
    }
    // online softmax (VALU + shuffles overlap the in-flight loads above)
#pragma unroll
    for (int vi = 0; vi < 8; ++vi) {
      float s0 = s[0][vi] * 0.125f + addm[0];
      float s1 = s[1][vi] * 0.125f + addm[1];
      float cm = fmaxf(s0, s1);
#pragma unroll
      for (int off = 1; off <= 8; off <<= 1) cm = fmaxf(cm, __shfl_xor(cm, off, 32));
      float nm    = fmaxf(rmax[vi], cm);
      float alpha = __expf(rmax[vi] - nm);
      rmax[vi] = nm;
      float e0 = __expf(s0 - nm);
      float e1 = __expf(s1 - nm);
      float ps = e0 + e1;
#pragma unroll
      for (int off = 1; off <= 8; off <<= 1) ps += __shfl_xor(ps, off, 32);
      rsum[vi] = rsum[vi] * alpha + ps;
#pragma unroll
      for (int t = 0; t < 4; ++t) acc[t][vi] *= alpha;
      int m = hl * 8 + vi;                  // D-layout row
      lds_p[w][m * 40 + n]      = (__bf16)e0;
      lds_p[w][m * 40 + 16 + n] = (__bf16)e1;
    }
    // Reload P in A-fragment layout (same-wave DS ops are in-order)
    v16bf pf;
    {
      const __bf16* lp = &lds_p[w][(lane & 15) * 40];
      v8bf r0 = *(const v8bf*)(lp + hl * 16);
      v8bf r1 = *(const v8bf*)(lp + hl * 16 + 8);
#pragma unroll
      for (int i = 0; i < 8; ++i) { pf[i] = r0[i]; pf[8 + i] = r1[i]; }
    }
#pragma unroll
    for (int t = 0; t < 4; ++t) acc[t] = wmma_bf16(pf, vf[t], acc[t]);
#pragma unroll
    for (int j = 0; j < 2; ++j) { kb[j][0] = kbn[j][0]; kb[j][1] = kbn[j][1]; mv[j] = mvn[j]; }
  }

#pragma unroll
  for (int vi = 0; vi < 8; ++vi) {
    float inv = 1.0f / rsum[vi];
    int m = hl * 8 + vi;
    int srow = q0 + m;
#pragma unroll
    for (int t = 0; t < 4; ++t) {
      Ctx[(long)(b * SEQ + srow) * HIDDEN + head * HDIM + t * 16 + n] =
          (__bf16)(acc[t][vi] * inv);
    }
  }
}

extern "C" void kernel_launch(void* const* d_in, const int* in_sizes, int n_in,
                              void* d_out, int out_size, void* d_ws, size_t ws_size,
                              hipStream_t stream) {
  const float* X    = (const float*)d_in[0];
  const float* mask = (const float*)d_in[1];
  const float* Wq = (const float*)d_in[2]; const float* bq = (const float*)d_in[3];
  const float* Wk = (const float*)d_in[4]; const float* bk = (const float*)d_in[5];
  const float* Wv = (const float*)d_in[6]; const float* bv = (const float*)d_in[7];
  const float* Wo = (const float*)d_in[8]; const float* bo = (const float*)d_in[9];
  float* out = (float*)d_out;

  char* ws = (char*)d_ws;
  __bf16* Xbf = (__bf16*)(ws);                       // 4096x1024        (8 MiB)
  __bf16* Wqt = (__bf16*)(ws + (8u  << 20));         // 4 x 1024x1024    (8 MiB)
  __bf16* Wkt = Wqt + 1048576;
  __bf16* Wvt = Wkt + 1048576;
  __bf16* Wot = Wvt + 1048576;
  __bf16* Qbf = (__bf16*)(ws + (16u << 20));         // [B][H][S][D]     (8 MiB)
  __bf16* Kbf = Qbf + 4194304;                       // [B][H][S][D]
  __bf16* Vtb = Kbf + 4194304;                       // [B][H][D][S]
  __bf16* Ctx = Vtb + 4194304;                       // [B*S][1024]

  cast_bf16_kernel<<<(MROWS * HIDDEN) / 256, 256, 0, stream>>>(X, Xbf, MROWS * HIDDEN);
  transpose_cast_kernel<<<4096, 256, 0, stream>>>(Wq, Wqt);
  transpose_cast_kernel<<<4096, 256, 0, stream>>>(Wk, Wkt);
  transpose_cast_kernel<<<4096, 256, 0, stream>>>(Wv, Wvt);
  transpose_cast_kernel<<<4096, 256, 0, stream>>>(Wo, Wot);

  // 64 m-blocks (64 rows each) x 16 n-tiles = 1024 blocks of 4 waves
  gemm_bf16_kernel<<<1024, 128, 0, stream>>>(Xbf, Wqt, bq, Qbf, 64, 0);
  gemm_bf16_kernel<<<1024, 128, 0, stream>>>(Xbf, Wkt, bk, Kbf, 64, 0);
  gemm_bf16_kernel<<<1024, 128, 0, stream>>>(Xbf, Wvt, bv, Vtb, 64, 1);

  attn_kernel<<<dim3(SEQ / 64, NHEADS, BATCH), 128, 0, stream>>>(Qbf, Kbf, Vtb, mask, Ctx);

  gemm_bf16_kernel<<<1024, 128, 0, stream>>>(Ctx, Wot, bo, out, 64, 2);
}
